// Synapformer_4037269258354
// MI455X (gfx1250) — matrix operand order repaired and checked by reading, not compiled, not run on hardware
//
#include <hip/hip_runtime.h>
#include <cstdint>

#define Gc      2048
#define P0c     32
#define S1c     16
#define S2c     4
#define NNODES  32768
#define EDGES   (Gc + NNODES)     // real edges + self loops
#define N1      (Gc * S1c)        // 32768
#define GS2v    (Gc * S2c)        // 8192
#define ROWS2   (GS2v * S1c)      // 131072

typedef __attribute__((ext_vector_type(16))) _Float16 v16h;
typedef __attribute__((ext_vector_type(8)))  float    v8f;

// ---------------- WMMA fragment helpers (CDNA5 16x16x32 f16, wave32) ----------------
// A 16x32 (MxK) from f32 row-major: lanes 0-15 row=lane, elems 0..7 -> K 0..7, 8..15 -> K 16..23
//                                   lanes 16-31 row=lane-16, elems 0..7 -> K 8..15, 8..15 -> K 24..31
// Requires K % 32 == 0 (rows 32B-aligned), M % 16 == 0.
__device__ inline v16h load_a_frag(const float* A, int lda, int row0, int k0) {
  int lane = threadIdx.x & 31;
  const float* Ar = A + (size_t)(row0 + (lane & 15)) * lda + (size_t)k0 + ((lane < 16) ? 0 : 8);
  float4 p0 = *(const float4*)(Ar + 0);
  float4 p1 = *(const float4*)(Ar + 4);
  float4 p2 = *(const float4*)(Ar + 16);
  float4 p3 = *(const float4*)(Ar + 20);
  v16h a;
  a[0]  = (_Float16)p0.x; a[1]  = (_Float16)p0.y; a[2]  = (_Float16)p0.z; a[3]  = (_Float16)p0.w;
  a[4]  = (_Float16)p1.x; a[5]  = (_Float16)p1.y; a[6]  = (_Float16)p1.z; a[7]  = (_Float16)p1.w;
  a[8]  = (_Float16)p2.x; a[9]  = (_Float16)p2.y; a[10] = (_Float16)p2.z; a[11] = (_Float16)p2.w;
  a[12] = (_Float16)p3.x; a[13] = (_Float16)p3.y; a[14] = (_Float16)p3.z; a[15] = (_Float16)p3.w;
  return a;
}
// B fragment from pre-converted fragment-major f16 weights:
// Wh layout = [K/32][N/16][lane 0..31][16 halfs]; one 32-byte contiguous read per lane.
__device__ inline v16h load_b_frag_h(const _Float16* Wh, int N, int k0, int col0) {
  int lane = threadIdx.x & 31;
  const _Float16* p = Wh + ((((size_t)(k0 >> 5) * (N >> 4)) + (col0 >> 4)) * 32 + lane) * 16;
  return *(const v16h*)p;
}
// C/D 16x16 f32: elem r -> row = row0 + r + (lane<16?0:8), col = col0 + (lane&15)
__device__ inline void store_c_frag(float* C, int ldc, int row0, int col0, v8f c,
                                    const float* bias, int act) {
  int lane = threadIdx.x & 31;
  int cN = col0 + (lane & 15);
  int rb = row0 + ((lane < 16) ? 0 : 8);
  float bv = bias ? bias[cN] : 0.f;
#pragma unroll
  for (int r = 0; r < 8; ++r) {
    float v = c[r] + bv;
    if (act == 1) v = fmaxf(v, 0.f);
    else if (act == 2) v = (v > 0.f) ? v : (__expf(v) - 1.f);  // elu
    C[(size_t)(rb + r) * ldc + cN] = v;
  }
}

// -------- weight convert f32 [K,N] -> padded f16 fragment-major [Kp/32][Np/16][32][16] --------
__global__ void k_wconv(const float* __restrict__ W, _Float16* __restrict__ Wh,
                        int K, int N, int Kp, int Np) {
  long idx = (long)blockIdx.x * blockDim.x + threadIdx.x;
  if (idx >= (long)Kp * Np) return;
  int k = (int)(idx / Np), n = (int)(idx % Np);
  float v = (k < K && n < N) ? W[(size_t)k * N + n] : 0.f;
  int lane = (n & 15) + ((k & 8) ? 16 : 0);
  int e = (k & 7) + ((k & 16) >> 1);
  size_t off = ((((size_t)(k >> 5) * (Np >> 4)) + (n >> 4)) * 32 + lane) * 16 + e;
  Wh[off] = (_Float16)v;
}

// ------------- generic GEMM: C[M,N] = act(A[M,K] @ W[K,N] + b) -------------
// Requirements: M%16==0, N%64==0, K%32==0. Per wave: 16x64 tile (4 wmma accumulators).
__global__ void __launch_bounds__(128) k_gemm(const float* __restrict__ A,
                                              const _Float16* __restrict__ Wh,
                                              const float* __restrict__ bias,
                                              float* __restrict__ C,
                                              int M, int N, int K, int act) {
  int wv = threadIdx.x >> 5;
  int nt = N >> 6;
  long t = (long)blockIdx.x * 4 + wv;
  if (t >= (long)(M >> 4) * nt) return;
  int tm = (int)(t / nt), tn = (int)(t % nt);
  int row0 = tm << 4, col0 = tn << 6;
  v8f c0 = {}, c1 = {}, c2 = {}, c3 = {};
  for (int k0 = 0; k0 < K; k0 += 32) {
    v16h a  = load_a_frag(A, K, row0, k0);
    v16h b0 = load_b_frag_h(Wh, N, k0, col0);
    v16h b1 = load_b_frag_h(Wh, N, k0, col0 + 16);
    v16h b2 = load_b_frag_h(Wh, N, k0, col0 + 32);
    v16h b3 = load_b_frag_h(Wh, N, k0, col0 + 48);
    c0 = __builtin_amdgcn_wmma_f32_16x16x32_f16(false, a, false, b0, (short)0, c0, false, false);
    c1 = __builtin_amdgcn_wmma_f32_16x16x32_f16(false, a, false, b1, (short)0, c1, false, false);
    c2 = __builtin_amdgcn_wmma_f32_16x16x32_f16(false, a, false, b2, (short)0, c2, false, false);
    c3 = __builtin_amdgcn_wmma_f32_16x16x32_f16(false, a, false, b3, (short)0, c3, false, false);
  }
  store_c_frag(C, N, row0, col0,      c0, bias, act);
  store_c_frag(C, N, row0, col0 + 16, c1, bias, act);
  store_c_frag(C, N, row0, col0 + 32, c2, bias, act);
  store_c_frag(C, N, row0, col0 + 48, c3, bias, act);
}

// ---------------- bias zero-padding (for the padded classifier head) ----------------
__global__ void k_padb(const float* __restrict__ b, float* __restrict__ bp, int N, int Np) {
  int i = blockIdx.x * blockDim.x + threadIdx.x;
  if (i >= Np) return;
  bp[i] = (i < N) ? b[i] : 0.f;
}

// ---------------- farthest point sampling (one cloud per block, wave32) ----------------
__global__ void k_fps(const float* __restrict__ pos, int P, int S,
                      int* __restrict__ idx_out, float* __restrict__ q_out) {
  int g = blockIdx.x;
  __shared__ float sp[32][6];
  __shared__ float sd[32];
  __shared__ int snxt;
  int lane = threadIdx.x;
  bool act = lane < P;
  if (act)
    for (int d = 0; d < 6; ++d) sp[lane][d] = pos[((size_t)g * P + lane) * 6 + d];
  __syncthreads();
  float dcur = -1e30f;
  if (act) {
    dcur = 0.f;
    for (int d = 0; d < 6; ++d) { float df = sp[lane][d] - sp[0][d]; dcur += df * df; }
  }
  if (lane == 0) {
    idx_out[(size_t)g * S] = 0;
    for (int d = 0; d < 6; ++d) q_out[((size_t)g * S) * 6 + d] = sp[0][d];
  }
  for (int i = 1; i < S; ++i) {
    sd[lane] = act ? dcur : -1e30f;
    __syncthreads();
    if (lane == 0) {
      int best = 0; float bv = sd[0];
      for (int j = 1; j < P; ++j) if (sd[j] > bv) { bv = sd[j]; best = j; }
      snxt = best;
      idx_out[(size_t)g * S + i] = best;
      for (int d = 0; d < 6; ++d) q_out[((size_t)g * S + i) * 6 + d] = sp[best][d];
    }
    __syncthreads();
    int nxt = snxt;
    if (act) {
      float dn = 0.f;
      for (int d = 0; d < 6; ++d) { float df = sp[lane][d] - sp[nxt][d]; dn += df * df; }
      dcur = fminf(dcur, dn);
    }
    __syncthreads();
  }
}

// ---------------- SA1 fused: per (g,s) query: rel -> MLP(6,64,64,128) -> masked max ----
// W*h are fragment-major f16 weights (layer1 zero-padded to K=32).
__global__ void __launch_bounds__(128) k_sa1(const float* __restrict__ syn,
                                             const float* __restrict__ q1,
                                             const _Float16* __restrict__ W1h, const float* __restrict__ b1,
                                             const _Float16* __restrict__ W2h, const float* __restrict__ b2,
                                             const _Float16* __restrict__ W3h, const float* __restrict__ b3,
                                             float* __restrict__ x1f) {
  int g = blockIdx.x >> 4;
  int s = blockIdx.x & 15;
  __shared__ float rel[32][32];
  __shared__ float h1[32][64];
  __shared__ float h2[32][64];
  __shared__ float h3[32][128];
  __shared__ unsigned mbits;
  int tid = threadIdx.x;
  if (tid < 32) {
    float d2 = 0.f;
#pragma unroll
    for (int d = 0; d < 6; ++d) {
      float r = syn[((size_t)g * 32 + tid) * 6 + d] - q1[((size_t)g * 16 + s) * 6 + d];
      rel[tid][d] = r; d2 += r * r;
    }
#pragma unroll
    for (int d = 6; d < 32; ++d) rel[tid][d] = 0.f;
    unsigned long long bal = __ballot(d2 < 0.04f);   // R1^2 = 0.2^2
    if (tid == 0) mbits = (unsigned)bal;
  }
  __syncthreads();
  int wv = tid >> 5;
  for (int t = wv; t < 8; t += 4) {                  // layer1 [32,32pad]@[32,64], relu
    int tm = t & 1, tn = t >> 1;
    v8f c = {};
    v16h a = load_a_frag(&rel[0][0], 32, tm * 16, 0);
    v16h b = load_b_frag_h(W1h, 64, 0, tn * 16);
    c = __builtin_amdgcn_wmma_f32_16x16x32_f16(false, a, false, b, (short)0, c, false, false);
    store_c_frag(&h1[0][0], 64, tm * 16, tn * 16, c, b1, 1);
  }
  __syncthreads();
  for (int t = wv; t < 8; t += 4) {                  // layer2 [32,64]@[64,64], relu
    int tm = t & 1, tn = t >> 1;
    v8f c = {};
    for (int k0 = 0; k0 < 64; k0 += 32) {
      v16h a = load_a_frag(&h1[0][0], 64, tm * 16, k0);
      v16h b = load_b_frag_h(W2h, 64, k0, tn * 16);
      c = __builtin_amdgcn_wmma_f32_16x16x32_f16(false, a, false, b, (short)0, c, false, false);
    }
    store_c_frag(&h2[0][0], 64, tm * 16, tn * 16, c, b2, 1);
  }
  __syncthreads();
  for (int t = wv; t < 16; t += 4) {                 // layer3 [32,64]@[64,128], no act
    int tm = t & 1, tn = t >> 1;
    v8f c = {};
    for (int k0 = 0; k0 < 64; k0 += 32) {
      v16h a = load_a_frag(&h2[0][0], 64, tm * 16, k0);
      v16h b = load_b_frag_h(W3h, 128, k0, tn * 16);
      c = __builtin_amdgcn_wmma_f32_16x16x32_f16(false, a, false, b, (short)0, c, false, false);
    }
    store_c_frag(&h3[0][0], 128, tm * 16, tn * 16, c, b3, 0);
  }
  __syncthreads();
  unsigned mk = mbits;
  float v = -3.0e38f;
  for (int p = 0; p < 32; ++p)
    if ((mk >> p) & 1) v = fmaxf(v, h3[p][tid]);
  x1f[(size_t)blockIdx.x * 128 + tid] = v;
}

// ---------------- SA2 self-loop constants: delta0 = pos_nn(0), e_self = attn(delta0) ----
__global__ void k_pt_self(const float* __restrict__ bp1, const float* __restrict__ Wp2,
                          const float* __restrict__ bp2, const float* __restrict__ Wa,
                          const float* __restrict__ ba,
                          float* __restrict__ delta0, float* __restrict__ e_self) {
  __shared__ float t64[64];
  __shared__ float d0s[256];
  int tid = threadIdx.x;
  if (tid < 64) t64[tid] = fmaxf(bp1[tid], 0.f);
  __syncthreads();
  float acc = bp2[tid];
  for (int k = 0; k < 64; ++k) acc += t64[k] * Wp2[k * 256 + tid];
  acc = fmaxf(acc, 0.f);
  d0s[tid] = acc; delta0[tid] = acc;
  __syncthreads();
  float e = ba[tid];
  for (int k = 0; k < 256; ++k) e += d0s[k] * Wa[k * 256 + tid];
  e_self[tid] = e;
}

// rel2 padded to 32 cols so the GEMM K-loop is branch-free
__global__ void k_rel2(const float* __restrict__ q1, float* __restrict__ rel2) {
  long idx = (long)blockIdx.x * blockDim.x + threadIdx.x;
  if (idx >= (long)ROWS2 * 32) return;
  int row = (int)(idx >> 5), d = (int)(idx & 31);
  int g = row >> 6, s2 = (row >> 4) & 3, s1 = row & 15;
  float v = 0.f;
  if (d < 6)
    v = q1[((size_t)g * 4 + s2) * 6 + d] - q1[((size_t)g * 16 + s1) * 6 + d];
  rel2[idx] = v;
}

__global__ void k_epre(const float* __restrict__ adst, const float* __restrict__ asrc,
                       const float* __restrict__ delta, float* __restrict__ epre) {
  long idx = (long)blockIdx.x * blockDim.x + threadIdx.x;
  if (idx >= (long)ROWS2 * 256) return;
  int row = (int)(idx >> 8), f = (int)(idx & 255);
  int g = row >> 6, s2 = (row >> 4) & 3, s1 = row & 15;
  epre[idx] = adst[((size_t)g * 4 + s2) * 256 + f] - asrc[((size_t)g * 16 + s1) * 256 + f] + delta[idx];
}

__global__ void k_sa2_softmax(const float* __restrict__ e, const float* __restrict__ delta,
                              const float* __restrict__ xl, const float* __restrict__ q1,
                              const float* __restrict__ q2, const float* __restrict__ delta0,
                              const float* __restrict__ e_self, float* __restrict__ out) {
  int b = blockIdx.x;          // b = g*4 + s2
  int g = b >> 2;
  __shared__ unsigned mbits;
  int tid = threadIdx.x;
  if (tid < 16) {
    float d2 = 0.f;
    for (int d = 0; d < 6; ++d) {
      float df = q1[((size_t)g * 16 + tid) * 6 + d] - q2[(size_t)b * 6 + d];
      d2 += df * df;
    }
    bool m = (d2 < 0.16f) && ((g * 16 + tid) != b);   // R2^2, remove_self_loops in flat space
    unsigned long long bal = __ballot(m);
    if (tid == 0) mbits = (unsigned)bal;
  }
  __syncthreads();
  unsigned mk = mbits;
  int f = tid;
  float m = e_self[f];
  for (int s1 = 0; s1 < 16; ++s1)
    if ((mk >> s1) & 1) m = fmaxf(m, e[((size_t)(b * 16 + s1)) * 256 + f]);
  float num = 0.f, den = 0.f;
  for (int s1 = 0; s1 < 16; ++s1)
    if ((mk >> s1) & 1) {
      size_t row = (size_t)b * 16 + s1;
      float w = __expf(e[row * 256 + f] - m);
      num += w * (xl[((size_t)g * 16 + s1) * 256 + f] + delta[row * 256 + f]);
      den += w;
    }
  float ws = __expf(e_self[f] - m);
  num += ws * (xl[(size_t)b * 256 + f] + delta0[f]);
  den += ws;
  out[(size_t)b * 256 + f] = num / den;
}

__global__ void k_sa2_rest(const float* __restrict__ xl, const float* __restrict__ delta0,
                           float* __restrict__ out) {
  long idx = (long)blockIdx.x * blockDim.x + threadIdx.x;
  long tot = (long)(N1 - GS2v) * 256;
  if (idx >= tot) return;
  long row = GS2v + idx / 256;
  int f = (int)(idx % 256);
  out[row * 256 + f] = xl[row * 256 + f] + delta0[f];
}

// concat padded to 288 cols (256 feats + 6 pos + 26 zeros)
__global__ void k_concat2(const float* __restrict__ a, const float* __restrict__ p6,
                          float* __restrict__ cat) {
  long idx = (long)blockIdx.x * blockDim.x + threadIdx.x;
  if (idx >= (long)N1 * 288) return;
  long row = idx / 288; int c = (int)(idx % 288);
  float v = 0.f;
  if (c < 256) v = a[row * 256 + c];
  else if (c < 262) v = p6[row * 6 + (c - 256)];
  cat[idx] = v;
}

__global__ void k_rowmax16(const float* __restrict__ x2, float* __restrict__ xg) {
  long idx = (long)blockIdx.x * blockDim.x + threadIdx.x;
  if (idx >= (long)Gc * 1024) return;
  long g = idx / 1024; int f = (int)(idx % 1024);
  float v = -3.0e38f;
  for (int s = 0; s < 16; ++s) v = fmaxf(v, x2[((size_t)g * 16 + s) * 1024 + f]);
  xg[idx] = v;
}

// ---------------- GAT helpers ----------------
__device__ inline unsigned encf(float f) {
  unsigned b = __float_as_uint(f);
  return (b & 0x80000000u) ? ~b : (b | 0x80000000u);
}
__device__ inline float decf(unsigned k) {
  return __uint_as_float((k & 0x80000000u) ? (k ^ 0x80000000u) : ~k);
}
__device__ inline void edge_sd(const int* ei, int ed, int& s, int& d) {
  if (ed < Gc) { s = ei[ed]; d = ei[Gc + ed]; }
  else { s = d = ed - Gc; }
}

__global__ void k_attn_coef(const float* __restrict__ h, const float* __restrict__ att,
                            float* __restrict__ out, int Ntot, int Hh, int Cc) {
  long idx = (long)blockIdx.x * blockDim.x + threadIdx.x;
  if (idx >= (long)Ntot * Hh) return;
  long n_ = idx / Hh; int hh = (int)(idx % Hh);
  const float* hp = h + (size_t)n_ * Hh * Cc + (size_t)hh * Cc;
  const float* ap = att + (size_t)hh * Cc;
  float acc = 0.f;
  for (int c = 0; c < Cc; ++c) acc += hp[c] * ap[c];
  out[idx] = acc;
}

__global__ void k_edge_logit(const float* __restrict__ as_, const float* __restrict__ ad_,
                             const int* __restrict__ ei, int Hh, float* __restrict__ e_out) {
  long idx = (long)blockIdx.x * blockDim.x + threadIdx.x;
  if (idx >= (long)EDGES * Hh) return;
  int ed = (int)(idx / Hh), hh = (int)(idx % Hh);
  int s, d; edge_sd(ei, ed, s, d);
  float v = as_[(size_t)s * Hh + hh] + ad_[(size_t)d * Hh + hh];
  e_out[idx] = (v > 0.f) ? v : 0.2f * v;   // leaky_relu(0.2)
}

__global__ void k_segmax(const float* __restrict__ e, const int* __restrict__ ei, int Hh,
                         unsigned* __restrict__ mx) {
  long idx = (long)blockIdx.x * blockDim.x + threadIdx.x;
  if (idx >= (long)EDGES * Hh) return;
  int ed = (int)(idx / Hh), hh = (int)(idx % Hh);
  int s, d; edge_sd(ei, ed, s, d); (void)s;
  atomicMax(&mx[(size_t)d * Hh + hh], encf(e[idx]));
}

__global__ void k_expsum(const float* __restrict__ e, const int* __restrict__ ei, int Hh,
                         const unsigned* __restrict__ mx, float* __restrict__ z,
                         float* __restrict__ wout) {
  long idx = (long)blockIdx.x * blockDim.x + threadIdx.x;
  if (idx >= (long)EDGES * Hh) return;
  int ed = (int)(idx / Hh), hh = (int)(idx % Hh);
  int s, d; edge_sd(ei, ed, s, d); (void)s;
  float w = __expf(e[idx] - decf(mx[(size_t)d * Hh + hh]));
  wout[idx] = w;
  atomicAdd(&z[(size_t)d * Hh + hh], w);
}

__global__ void k_accum(const float* __restrict__ wv, const float* __restrict__ z,
                        const float* __restrict__ h, const int* __restrict__ ei,
                        int Hh, int Cc, float* __restrict__ agg) {
  long idx = (long)blockIdx.x * blockDim.x + threadIdx.x;
  long tot = (long)EDGES * Hh * Cc;
  if (idx >= tot) return;
  int ed = (int)(idx / (Hh * Cc));
  int r = (int)(idx % (Hh * Cc));
  int hh = r / Cc;
  int s, d; edge_sd(ei, ed, s, d);
  float alpha = wv[(size_t)ed * Hh + hh] / z[(size_t)d * Hh + hh];
  atomicAdd(&agg[(size_t)d * Hh * Cc + r], h[(size_t)s * Hh * Cc + r] * alpha);
}

__global__ void k_bias_act(const float* __restrict__ in, const float* __restrict__ bias,
                           float* __restrict__ out, long total, int F, int act) {
  long idx = (long)blockIdx.x * blockDim.x + threadIdx.x;
  if (idx >= total) return;
  float v = in[idx] + bias[idx % F];
  if (act == 2) v = (v > 0.f) ? v : (__expf(v) - 1.f);
  out[idx] = v;
}

__global__ void k_fill_f(float* p, long n, float v) {
  long i = (long)blockIdx.x * blockDim.x + threadIdx.x;
  if (i < n) p[i] = v;
}
__global__ void k_fill_u(unsigned* p, long n, unsigned v) {
  long i = (long)blockIdx.x * blockDim.x + threadIdx.x;
  if (i < n) p[i] = v;
}

__global__ void k_edge_cnt(const int* __restrict__ ei, int* __restrict__ cntL,
                           int* __restrict__ cntR) {
  int ed = blockIdx.x * blockDim.x + threadIdx.x;
  if (ed >= Gc) return;
  atomicAdd(&cntL[ei[ed]], 1);
  atomicAdd(&cntR[ei[Gc + ed]], 1);
}

__global__ void k_scat_enc(const float* __restrict__ xp, const int* __restrict__ ei,
                           unsigned* __restrict__ encL, unsigned* __restrict__ encR) {
  long idx = (long)blockIdx.x * blockDim.x + threadIdx.x;
  if (idx >= (long)Gc * 512) return;
  int ed = (int)(idx >> 9), f = (int)(idx & 511);
  unsigned key = encf(xp[idx]);
  atomicMax(&encL[(size_t)ei[ed] * 512 + f], key);
  atomicMax(&encR[(size_t)ei[Gc + ed] * 512 + f], key);
}

__global__ void k_final(const float* __restrict__ xg2, const unsigned* __restrict__ encL,
                        const int* __restrict__ cntL, const unsigned* __restrict__ encR,
                        const int* __restrict__ cntR, float* __restrict__ fin) {
  long idx = (long)blockIdx.x * blockDim.x + threadIdx.x;
  if (idx >= (long)NNODES * 1280) return;
  long n_ = idx / 1280; int c = (int)(idx % 1280);
  float v;
  if (c < 256) v = xg2[n_ * 256 + c];
  else if (c < 768) v = (cntL[n_] > 0) ? decf(encL[n_ * 512 + (c - 256)]) : 0.f;
  else v = (cntR[n_] > 0) ? decf(encR[n_ * 512 + (c - 768)]) : 0.f;
  fin[idx] = v;
}

// copy 10 valid columns out of padded 64-wide classifier output
__global__ void k_copy10(const float* __restrict__ cpad, float* __restrict__ out) {
  long idx = (long)blockIdx.x * blockDim.x + threadIdx.x;
  if (idx >= (long)NNODES * 10) return;
  long row = idx / 10; int c = (int)(idx % 10);
  out[idx] = cpad[row * 64 + c];
}

// ---------------- host side ----------------
static int cdivi(long a, long b) { return (int)((a + b - 1) / b); }

static void gemm(hipStream_t st, const float* A, const _Float16* Wh, const float* bias,
                 float* C, int M, int N, int K, int act) {
  long tiles = (long)(M / 16) * (N / 64);
  k_gemm<<<cdivi(tiles, 4), 128, 0, st>>>(A, Wh, bias, C, M, N, K, act);
}

static void wconv(hipStream_t st, const float* W, _Float16* Wh, int K, int N, int Kp, int Np) {
  k_wconv<<<cdivi((long)Kp * Np, 256), 256, 0, st>>>(W, Wh, K, N, Kp, Np);
}

extern "C" void kernel_launch(void* const* d_in, const int* in_sizes, int n_in,
                              void* d_out, int out_size, void* d_ws, size_t ws_size,
                              hipStream_t stream) {
  (void)in_sizes; (void)n_in; (void)out_size; (void)ws_size;
  // setup_inputs() insertion-order flattening
  const float* xin   = (const float*)d_in[0];
  const float* g1W   = (const float*)d_in[1];
  const float* g1as  = (const float*)d_in[2];
  const float* g1ad  = (const float*)d_in[3];
  const float* g1b   = (const float*)d_in[4];
  const float* g2W   = (const float*)d_in[5];
  const float* g2as  = (const float*)d_in[6];
  const float* g2ad  = (const float*)d_in[7];
  const float* g2b   = (const float*)d_in[8];
  const float* s1W0  = (const float*)d_in[9];
  const float* s1b0  = (const float*)d_in[10];
  const float* s1W1  = (const float*)d_in[11];
  const float* s1b1  = (const float*)d_in[12];
  const float* s1W2  = (const float*)d_in[13];
  const float* s1b2  = (const float*)d_in[14];
  const float* ptlin = (const float*)d_in[15];
  const float* ptsrc = (const float*)d_in[16];
  const float* ptdst = (const float*)d_in[17];
  const float* pn0W  = (const float*)d_in[18];
  const float* pn0b  = (const float*)d_in[19];
  const float* pn1W  = (const float*)d_in[20];
  const float* pn1b  = (const float*)d_in[21];
  const float* attW  = (const float*)d_in[22];
  const float* attb  = (const float*)d_in[23];
  const float* n2W0  = (const float*)d_in[24];
  const float* n2b0  = (const float*)d_in[25];
  const float* n2W1  = (const float*)d_in[26];
  const float* n2b1  = (const float*)d_in[27];
  const float* n2W2  = (const float*)d_in[28];
  const float* n2b2  = (const float*)d_in[29];
  const float* sfW0  = (const float*)d_in[30];
  const float* sfb0  = (const float*)d_in[31];
  const float* sfW1  = (const float*)d_in[32];
  const float* sfb1  = (const float*)d_in[33];
  const float* clW0  = (const float*)d_in[34];
  const float* clb0  = (const float*)d_in[35];
  const float* clW1  = (const float*)d_in[36];
  const float* clb1  = (const float*)d_in[37];
  const float* syn   = (const float*)d_in[38];
  const int*   eidx  = (const int*)d_in[39];
  float* out = (float*)d_out;

  char* wsp = (char*)d_ws;
  auto alloc = [&](size_t bytes) {
    void* p = (void*)wsp; wsp += (bytes + 255) & ~(size_t)255; return p;
  };
  float* q1    = (float*)alloc((size_t)N1 * 6 * 4);    // == pos1f
  int*   idx1  = (int*)alloc((size_t)N1 * 4);
  float* q2    = (float*)alloc((size_t)GS2v * 6 * 4);
  int*   idx2  = (int*)alloc((size_t)GS2v * 4);
  // f16 fragment-major weights
  _Float16* s1W0h = (_Float16*)alloc((size_t)32 * 64 * 2);
  _Float16* s1W1h = (_Float16*)alloc((size_t)64 * 64 * 2);
  _Float16* s1W2h = (_Float16*)alloc((size_t)64 * 128 * 2);
  _Float16* ptlinh = (_Float16*)alloc((size_t)128 * 256 * 2);
  _Float16* ptsrch = (_Float16*)alloc((size_t)128 * 256 * 2);
  _Float16* ptdsth = (_Float16*)alloc((size_t)128 * 256 * 2);
  _Float16* pn0Wh = (_Float16*)alloc((size_t)32 * 64 * 2);
  _Float16* pn1Wh = (_Float16*)alloc((size_t)64 * 256 * 2);
  _Float16* attWh = (_Float16*)alloc((size_t)256 * 256 * 2);
  _Float16* n2W0h = (_Float16*)alloc((size_t)288 * 256 * 2);
  _Float16* n2W1h = (_Float16*)alloc((size_t)256 * 512 * 2);
  _Float16* n2W2h = (_Float16*)alloc((size_t)512 * 1024 * 2);
  _Float16* sfW0h = (_Float16*)alloc((size_t)1024 * 512 * 2);
  _Float16* sfW1h = (_Float16*)alloc((size_t)512 * 512 * 2);
  _Float16* g1Wh  = (_Float16*)alloc((size_t)128 * 512 * 2);
  _Float16* g2Wh  = (_Float16*)alloc((size_t)512 * 256 * 2);
  _Float16* clW0h = (_Float16*)alloc((size_t)1280 * 512 * 2);
  _Float16* clW1h = (_Float16*)alloc((size_t)512 * 64 * 2);
  float* clb1p = (float*)alloc((size_t)64 * 4);
  // activations / scratch
  float* x1f   = (float*)alloc((size_t)N1 * 128 * 4);
  float* xl    = (float*)alloc((size_t)N1 * 256 * 4);
  float* asrc  = (float*)alloc((size_t)N1 * 256 * 4);
  float* adst  = (float*)alloc((size_t)N1 * 256 * 4);
  float* del0  = (float*)alloc(256 * 4);
  float* esel  = (float*)alloc(256 * 4);
  float* rel2  = (float*)alloc((size_t)ROWS2 * 32 * 4);
  float* t1    = (float*)alloc((size_t)ROWS2 * 64 * 4);
  float* delt  = (float*)alloc((size_t)ROWS2 * 256 * 4);
  float* epre  = (float*)alloc((size_t)ROWS2 * 256 * 4);
  float* ebuf  = (float*)alloc((size_t)ROWS2 * 256 * 4);
  float* outb  = (float*)alloc((size_t)N1 * 256 * 4);
  float* cat   = (float*)alloc((size_t)N1 * 288 * 4);
  float* n2a   = (float*)alloc((size_t)N1 * 256 * 4);
  float* n2bv  = (float*)alloc((size_t)N1 * 512 * 4);
  float* x2    = (float*)alloc((size_t)N1 * 1024 * 4);
  float* xg    = (float*)alloc((size_t)Gc * 1024 * 4);
  float* sfa   = (float*)alloc((size_t)Gc * 512 * 4);
  float* xpt   = (float*)alloc((size_t)Gc * 512 * 4);
  float* h1g   = (float*)alloc((size_t)NNODES * 512 * 4);
  float* as1   = (float*)alloc((size_t)NNODES * 8 * 4);
  float* ad1   = (float*)alloc((size_t)NNODES * 8 * 4);
  float* e1    = (float*)alloc((size_t)EDGES * 8 * 4);
  float* w1e   = (float*)alloc((size_t)EDGES * 8 * 4);
  unsigned* mx1 = (unsigned*)alloc((size_t)NNODES * 8 * 4);
  float* z1    = (float*)alloc((size_t)NNODES * 8 * 4);
  float* agg1  = (float*)alloc((size_t)NNODES * 512 * 4);
  float* xg1   = (float*)alloc((size_t)NNODES * 512 * 4);
  float* h2g   = (float*)alloc((size_t)NNODES * 256 * 4);
  float* as2   = (float*)alloc((size_t)NNODES * 4);
  float* ad2   = (float*)alloc((size_t)NNODES * 4);
  float* e2    = (float*)alloc((size_t)EDGES * 4);
  float* w2e   = (float*)alloc((size_t)EDGES * 4);
  unsigned* mx2 = (unsigned*)alloc((size_t)NNODES * 4);
  float* z2    = (float*)alloc((size_t)NNODES * 4);
  float* agg2  = (float*)alloc((size_t)NNODES * 256 * 4);
  float* xg2   = (float*)alloc((size_t)NNODES * 256 * 4);
  unsigned* encL = (unsigned*)alloc((size_t)NNODES * 512 * 4);
  unsigned* encR = (unsigned*)alloc((size_t)NNODES * 512 * 4);
  int* cntL  = (int*)alloc((size_t)NNODES * 4);
  int* cntR  = (int*)alloc((size_t)NNODES * 4);
  float* fin = (float*)alloc((size_t)NNODES * 1280 * 4);
  float* clt = (float*)alloc((size_t)NNODES * 512 * 4);
  float* clo = (float*)alloc((size_t)NNODES * 64 * 4);

  // ---- weight conversion to fragment-major f16 (once per launch, ~5 MB total) ----
  wconv(stream, s1W0, s1W0h, 6, 64, 32, 64);
  wconv(stream, s1W1, s1W1h, 64, 64, 64, 64);
  wconv(stream, s1W2, s1W2h, 64, 128, 64, 128);
  wconv(stream, ptlin, ptlinh, 128, 256, 128, 256);
  wconv(stream, ptsrc, ptsrch, 128, 256, 128, 256);
  wconv(stream, ptdst, ptdsth, 128, 256, 128, 256);
  wconv(stream, pn0W, pn0Wh, 6, 64, 32, 64);
  wconv(stream, pn1W, pn1Wh, 64, 256, 64, 256);
  wconv(stream, attW, attWh, 256, 256, 256, 256);
  wconv(stream, n2W0, n2W0h, 262, 256, 288, 256);
  wconv(stream, n2W1, n2W1h, 256, 512, 256, 512);
  wconv(stream, n2W2, n2W2h, 512, 1024, 512, 1024);
  wconv(stream, sfW0, sfW0h, 1024, 512, 1024, 512);
  wconv(stream, sfW1, sfW1h, 512, 512, 512, 512);
  wconv(stream, g1W, g1Wh, 128, 512, 128, 512);
  wconv(stream, g2W, g2Wh, 512, 256, 512, 256);
  wconv(stream, clW0, clW0h, 1280, 512, 1280, 512);
  wconv(stream, clW1, clW1h, 512, 10, 512, 64);
  k_padb<<<1, 64, 0, stream>>>(clb1, clb1p, 10, 64);

  // ---- SA1 ----
  k_fps<<<Gc, 32, 0, stream>>>(syn, P0c, S1c, idx1, q1);
  k_fps<<<Gc, 32, 0, stream>>>(q1, S1c, S2c, idx2, q2);
  k_sa1<<<N1, 128, 0, stream>>>(syn, q1, s1W0h, s1b0, s1W1h, s1b1, s1W2h, s1b2, x1f);

  // ---- SA2 (PointTransformerConv) ----
  gemm(stream, x1f, ptlinh, nullptr, xl,   N1, 256, 128, 0);
  gemm(stream, x1f, ptsrch, nullptr, asrc, N1, 256, 128, 0);
  gemm(stream, x1f, ptdsth, nullptr, adst, N1, 256, 128, 0);
  k_pt_self<<<1, 256, 0, stream>>>(pn0b, pn1W, pn1b, attW, attb, del0, esel);
  k_rel2<<<cdivi((long)ROWS2 * 32, 256), 256, 0, stream>>>(q1, rel2);
  gemm(stream, rel2, pn0Wh, pn0b, t1,   ROWS2, 64, 32, 1);
  gemm(stream, t1,   pn1Wh, pn1b, delt, ROWS2, 256, 64, 1);
  k_epre<<<cdivi((long)ROWS2 * 256, 256), 256, 0, stream>>>(adst, asrc, delt, epre);
  gemm(stream, epre, attWh, attb, ebuf, ROWS2, 256, 256, 0);
  k_sa2_softmax<<<GS2v, 256, 0, stream>>>(ebuf, delt, xl, q1, q2, del0, esel, outb);
  k_sa2_rest<<<cdivi((long)(N1 - GS2v) * 256, 256), 256, 0, stream>>>(xl, del0, outb);

  // ---- nn2 + pooling + sf_mlp ----
  k_concat2<<<cdivi((long)N1 * 288, 256), 256, 0, stream>>>(outb, q1, cat);
  gemm(stream, cat,  n2W0h, n2b0, n2a,  N1, 256, 288, 1);
  gemm(stream, n2a,  n2W1h, n2b1, n2bv, N1, 512, 256, 1);
  gemm(stream, n2bv, n2W2h, n2b2, x2,   N1, 1024, 512, 0);
  k_rowmax16<<<cdivi((long)Gc * 1024, 256), 256, 0, stream>>>(x2, xg);
  gemm(stream, xg,  sfW0h, sfb0, sfa, Gc, 512, 1024, 1);
  gemm(stream, sfa, sfW1h, sfb1, xpt, Gc, 512, 512, 0);

  // ---- GAT layer 1 (H=8, C=64, concat) ----
  gemm(stream, xin, g1Wh, nullptr, h1g, NNODES, 512, 128, 0);
  k_attn_coef<<<cdivi((long)NNODES * 8, 256), 256, 0, stream>>>(h1g, g1as, as1, NNODES, 8, 64);
  k_attn_coef<<<cdivi((long)NNODES * 8, 256), 256, 0, stream>>>(h1g, g1ad, ad1, NNODES, 8, 64);
  k_fill_u<<<cdivi((long)NNODES * 8, 256), 256, 0, stream>>>(mx1, (long)NNODES * 8, 0u);
  k_fill_f<<<cdivi((long)NNODES * 8, 256), 256, 0, stream>>>(z1, (long)NNODES * 8, 0.f);
  k_fill_f<<<cdivi((long)NNODES * 512, 256), 256, 0, stream>>>(agg1, (long)NNODES * 512, 0.f);
  k_edge_logit<<<cdivi((long)EDGES * 8, 256), 256, 0, stream>>>(as1, ad1, eidx, 8, e1);
  k_segmax<<<cdivi((long)EDGES * 8, 256), 256, 0, stream>>>(e1, eidx, 8, mx1);
  k_expsum<<<cdivi((long)EDGES * 8, 256), 256, 0, stream>>>(e1, eidx, 8, mx1, z1, w1e);
  k_accum<<<cdivi((long)EDGES * 512, 256), 256, 0, stream>>>(w1e, z1, h1g, eidx, 8, 64, agg1);
  k_bias_act<<<cdivi((long)NNODES * 512, 256), 256, 0, stream>>>(agg1, g1b, xg1,
                                                                 (long)NNODES * 512, 512, 2);
  // ---- GAT layer 2 (H=1, C=256, mean==identity) ----
  gemm(stream, xg1, g2Wh, nullptr, h2g, NNODES, 256, 512, 0);
  k_attn_coef<<<cdivi((long)NNODES, 256), 256, 0, stream>>>(h2g, g2as, as2, NNODES, 1, 256);
  k_attn_coef<<<cdivi((long)NNODES, 256), 256, 0, stream>>>(h2g, g2ad, ad2, NNODES, 1, 256);
  k_fill_u<<<cdivi((long)NNODES, 256), 256, 0, stream>>>(mx2, (long)NNODES, 0u);
  k_fill_f<<<cdivi((long)NNODES, 256), 256, 0, stream>>>(z2, (long)NNODES, 0.f);
  k_fill_f<<<cdivi((long)NNODES * 256, 256), 256, 0, stream>>>(agg2, (long)NNODES * 256, 0.f);
  k_edge_logit<<<cdivi((long)EDGES, 256), 256, 0, stream>>>(as2, ad2, eidx, 1, e2);
  k_segmax<<<cdivi((long)EDGES, 256), 256, 0, stream>>>(e2, eidx, 1, mx2);
  k_expsum<<<cdivi((long)EDGES, 256), 256, 0, stream>>>(e2, eidx, 1, mx2, z2, w2e);
  k_accum<<<cdivi((long)EDGES * 256, 256), 256, 0, stream>>>(w2e, z2, h2g, eidx, 1, 256, agg2);
  k_bias_act<<<cdivi((long)NNODES * 256, 256), 256, 0, stream>>>(agg2, g2b, xg2,
                                                                 (long)NNODES * 256, 256, 0);

  // ---- scatter-max synapse features onto edge endpoints + classifier ----
  k_fill_u<<<cdivi((long)NNODES * 512, 256), 256, 0, stream>>>(encL, (long)NNODES * 512, 0u);
  k_fill_u<<<cdivi((long)NNODES * 512, 256), 256, 0, stream>>>(encR, (long)NNODES * 512, 0u);
  k_fill_u<<<cdivi((long)NNODES, 256), 256, 0, stream>>>((unsigned*)cntL, (long)NNODES, 0u);
  k_fill_u<<<cdivi((long)NNODES, 256), 256, 0, stream>>>((unsigned*)cntR, (long)NNODES, 0u);
  k_edge_cnt<<<cdivi(Gc, 256), 256, 0, stream>>>(eidx, cntL, cntR);
  k_scat_enc<<<cdivi((long)Gc * 512, 256), 256, 0, stream>>>(xpt, eidx, encL, encR);
  k_final<<<cdivi((long)NNODES * 1280, 256), 256, 0, stream>>>(xg2, encL, cntL, encR, cntR, fin);
  gemm(stream, fin, clW0h, clb0, clt, NNODES, 512, 1280, 1);
  gemm(stream, clt, clW1h, clb1p, clo, NNODES, 64, 512, 0);
  k_copy10<<<cdivi((long)NNODES * 10, 256), 256, 0, stream>>>(clo, out);
}
